// MultiHeadAttention_64811056496662
// MI455X (gfx1250) — matrix operand order repaired and checked by reading, not compiled, not run on hardware
//
#include <hip/hip_runtime.h>
#include <hip/hip_bf16.h>

// ---------------------------------------------------------------------------
// MHA for gfx1250 (MI455X): all matmuls via v_wmma_f32_16x16x32_bf16 (wave32).
//   H=16 heads, D=64 head dim, B=4 batch, T=2048 tokens, N_DIM=1024.
// Round 3: softmax row-max reduction moved off the DS pipe onto VALU DPP16
// butterflies (quad_perm/row_half_mirror/row_mirror); all 4 K fragments
// loaded up-front into distinct registers so one clause covers 8x
// global_load_b128 and WMMAs overlap the load latency.
// ---------------------------------------------------------------------------

typedef __attribute__((ext_vector_type(16))) __bf16 v16bf;
typedef __attribute__((ext_vector_type(8)))  __bf16 v8bf;
typedef __attribute__((ext_vector_type(8)))  float  v8f;

#define NH   16
#define BATCH 4
#define SEQ  2048
#define HD   64
#define ND   1024
#define MROWS (BATCH * SEQ)   // 8192

static __device__ __forceinline__ unsigned short f2bu(float f) {
    unsigned u = __builtin_bit_cast(unsigned, f);
    u += 0x7FFFu + ((u >> 16) & 1u);      // round-to-nearest-even
    return (unsigned short)(u >> 16);
}
static __device__ __forceinline__ __bf16 f2b(float f) {
    unsigned short s = f2bu(f);
    return __builtin_bit_cast(__bf16, s);
}
static __device__ __forceinline__ unsigned short bbits(__bf16 h) {
    return __builtin_bit_cast(unsigned short, h);
}
static __device__ __forceinline__ v16bf join16(v8bf lo, v8bf hi) {
    return __builtin_shufflevector(lo, hi, 0,1,2,3,4,5,6,7,8,9,10,11,12,13,14,15);
}

// max with a DPP-permuted copy of x (dpp_ctrl immediate), f32 via bitcast
template <int CTRL>
static __device__ __forceinline__ float max_dpp(float x) {
    int xi = __builtin_bit_cast(int, x);
    int pi = __builtin_amdgcn_update_dpp(xi, xi, CTRL, 0xF, 0xF, true);
    return fmaxf(x, __builtin_bit_cast(float, pi));
}
// full max over each 16-lane half: xor1, xor2, xor7(half-mirror), xor15(mirror)
static __device__ __forceinline__ float rowmax16(float x) {
    x = max_dpp<0xB1>(x);    // quad_perm(1,0,3,2)
    x = max_dpp<0x4E>(x);    // quad_perm(2,3,0,1)
    x = max_dpp<0x141>(x);   // row_half_mirror
    x = max_dpp<0x140>(x);   // row_mirror
    return x;
}

// ---------------------------------------------------------------------------
// GEMM: Out[m,n] = sum_k A[m,k] * W[n,k] + bias[n]
//   A: [8192,1024] (fp32 if A_F32, else bf16), W: [1024,1024] fp32 row-major.
//   OUT_HEADS: scatter bf16 into [h][b][t][64]; else fp32 [m][1024].
// Block = 128 threads (4 waves). Tile 64x64, K-step 32.
// ---------------------------------------------------------------------------
template <bool A_F32, bool OUT_HEADS>
__global__ __launch_bounds__(128) void gemm_wmma(const void* __restrict__ Aptr,
                                                 const float* __restrict__ W,
                                                 const float* __restrict__ bias,
                                                 void* __restrict__ Out) {
    __shared__ unsigned short sA[64 * 32];
    __shared__ unsigned short sB[64 * 32];

    const int tid  = threadIdx.x;
    const int wv   = tid >> 5;
    const int lane = tid & 31;
    const int m0   = blockIdx.y * 64;
    const int n0   = blockIdx.x * 64;

    v8f acc[4];
    #pragma unroll
    for (int i = 0; i < 4; ++i) acc[i] = (v8f){0.f,0.f,0.f,0.f,0.f,0.f,0.f,0.f};

    const int srow = tid >> 1;          // 0..63
    const int sk0  = (tid & 1) * 16;    // 0 or 16

    for (int kc = 0; kc < ND; kc += 32) {
        __syncthreads();
        // ---- stage A tile (64x32) as bf16 into LDS ----
        {
            unsigned short tmp[16];
            if (A_F32) {
                const float4* gp = reinterpret_cast<const float4*>(
                    (const float*)Aptr + (size_t)(m0 + srow) * ND + kc + sk0);
                #pragma unroll
                for (int j = 0; j < 4; ++j) {
                    float4 f = gp[j];
                    tmp[4*j+0] = f2bu(f.x); tmp[4*j+1] = f2bu(f.y);
                    tmp[4*j+2] = f2bu(f.z); tmp[4*j+3] = f2bu(f.w);
                }
            } else {
                const v16bf* gp = reinterpret_cast<const v16bf*>(
                    (const __bf16*)Aptr + (size_t)(m0 + srow) * ND + kc + sk0);
                v16bf v = *gp;
                #pragma unroll
                for (int i = 0; i < 16; ++i) tmp[i] = bbits(v[i]);
            }
            unsigned* dst = reinterpret_cast<unsigned*>(&sA[srow * 32 + sk0]);
            #pragma unroll
            for (int i = 0; i < 8; ++i)
                dst[i] = (unsigned)tmp[2*i] | ((unsigned)tmp[2*i+1] << 16);
        }
        // ---- stage W tile (64x32) as bf16 into LDS ----
        {
            const float4* gp = reinterpret_cast<const float4*>(
                W + (size_t)(n0 + srow) * ND + kc + sk0);
            unsigned short tmp[16];
            #pragma unroll
            for (int j = 0; j < 4; ++j) {
                float4 f = gp[j];
                tmp[4*j+0] = f2bu(f.x); tmp[4*j+1] = f2bu(f.y);
                tmp[4*j+2] = f2bu(f.z); tmp[4*j+3] = f2bu(f.w);
            }
            unsigned* dst = reinterpret_cast<unsigned*>(&sB[srow * 32 + sk0]);
            #pragma unroll
            for (int i = 0; i < 8; ++i)
                dst[i] = (unsigned)tmp[2*i] | ((unsigned)tmp[2*i+1] << 16);
        }
        __syncthreads();

        // ---- A fragment: lane row = wv*16 + lane%16; K split per lane half ----
        const int mrl  = wv * 16 + (lane & 15);
        const int aoff = (lane & 16) ? 16 : 0;
        v16bf alo = *reinterpret_cast<const v16bf*>(&sA[mrl * 32 + aoff]);
        v16bf afrag = alo;

        const int boff = (lane & 16) ? 16 : 0;
        #pragma unroll
        for (int ns = 0; ns < 4; ++ns) {
            const int nl = ns * 16 + (lane & 15);
            v16bf bfrag = *reinterpret_cast<const v16bf*>(&sB[nl * 32 + boff]);
            acc[ns] = __builtin_amdgcn_wmma_f32_16x16x32_bf16(
                false, afrag, false, bfrag, (short)0, acc[ns], false, false);
        }
    }

    // ---- epilogue: bias + store ----
    #pragma unroll
    for (int ns = 0; ns < 4; ++ns) {
        const int n  = n0 + ns * 16 + (lane & 15);
        const float bvl = bias[n];
        #pragma unroll
        for (int r = 0; r < 8; ++r) {
            const int mg  = m0 + wv * 16 + ((lane & 16) ? r + 8 : r);
            const float v = acc[ns][r] + bvl;
            if (OUT_HEADS) {
                const int h = n >> 6, d = n & 63;
                const int b = mg >> 11, t = mg & (SEQ - 1);
                ((__bf16*)Out)[(((size_t)(h * BATCH + b) * SEQ + t) << 6) + d] = f2b(v);
            } else {
                ((float*)Out)[(size_t)mg * ND + n] = v;
            }
        }
    }
}

// ---------------------------------------------------------------------------
// Flash attention: block = (h, b, 64 query rows); 4 waves x 16 rows each.
// Keys streamed in chunks of 32. S via WMMA (K frags direct from global,
// all 4 loaded up-front), online softmax: row max via DPP16 butterflies,
// row SUM via WMMA against a ones B-fragment. P through LDS (C->A layout),
// V chunk staged transposed in LDS, O += P.V via WMMA.
// No 1/sqrt(d) scale (matches reference).
// ---------------------------------------------------------------------------
__global__ __launch_bounds__(128) void attn_wmma(const __bf16* __restrict__ q,
                                                 const __bf16* __restrict__ k,
                                                 const __bf16* __restrict__ v,
                                                 __bf16* __restrict__ o) {
    __shared__ unsigned short sVt[64 * 32];        // [d][key]
    __shared__ unsigned short sP[4][16 * 32];      // per-wave P tile [m][key]

    const int tid  = threadIdx.x;
    const int wv   = tid >> 5;
    const int lane = tid & 31;
    const int qt   = blockIdx.x;     // 0..31
    const int b    = blockIdx.y;     // 0..3
    const int h    = blockIdx.z;     // 0..15

    const int hb = h * BATCH + b;
    const __bf16* qbase = q + ((size_t)hb * SEQ) * HD;
    const __bf16* kbase = k + ((size_t)hb * SEQ) * HD;
    const __bf16* vbase = v + ((size_t)hb * SEQ) * HD;

    // preload Q A-fragments for the two 32-wide d chunks
    const int myq  = qt * 64 + wv * 16 + (lane & 15);
    const int aoff = (lane & 16) ? 16 : 0;
    v16bf aq[2];
    #pragma unroll
    for (int c = 0; c < 2; ++c) {
        const __bf16* qp = qbase + (size_t)myq * HD + c * 32;
        v16bf lo = *reinterpret_cast<const v16bf*>(qp + aoff);
        aq[c] = lo;
    }

    // ones B-fragment (bf16 1.0 in every slot) for WMMA row sums
    v16bf ones;
    #pragma unroll
    for (int i = 0; i < 16; ++i)
        ones[i] = __builtin_bit_cast(__bf16, (unsigned short)0x3F80);

    v8f oacc[4];
    #pragma unroll
    for (int i = 0; i < 4; ++i) oacc[i] = (v8f){0.f,0.f,0.f,0.f,0.f,0.f,0.f,0.f};
    float mrow[8], lrow[8], sc8[8];
    #pragma unroll
    for (int r = 0; r < 8; ++r) { mrow[r] = -3.0e38f; lrow[r] = 0.f; }

    const int kd0  = (lane & 16) ? 16 : 0;   // K frag d-offset within 32-chunk
    const int klane = lane & 15;

    for (int kv0 = 0; kv0 < SEQ; kv0 += 32) {
        // prefetch next chunk's K and V lines (global_prefetch_b8)
        if (kv0 + 32 < SEQ) {
            const int pk = kv0 + 32 + lane;
            __builtin_prefetch(kbase + (size_t)pk * HD, 0, 3);
            __builtin_prefetch(vbase + (size_t)pk * HD, 0, 3);
        }

        // ---- load all 4 K B-fragments up-front (distinct regs, one clause) ----
        const __bf16* krow0 = kbase + (size_t)(kv0 + klane) * HD;
        const __bf16* krow1 = kbase + (size_t)(kv0 + 16 + klane) * HD;
        v16bf kb00 = *reinterpret_cast<const v16bf*>(krow0 + kd0);        // c=0, sub0
        v16bf kb01 = *reinterpret_cast<const v16bf*>(krow1 + kd0);        // c=0, sub1
        v16bf kb10 = *reinterpret_cast<const v16bf*>(krow0 + 32 + kd0);   // c=1, sub0
        v16bf kb11 = *reinterpret_cast<const v16bf*>(krow1 + 32 + kd0);   // c=1, sub1

        __syncthreads();   // previous chunk's sVt reads complete
        // ---- stage V chunk transposed: sVt[d*32 + key] = V[kv0+key][d] ----
        {
            const int key  = tid & 31;
            const int d0   = (tid >> 5) * 16;
            v16bf vv = *reinterpret_cast<const v16bf*>(
                vbase + (size_t)(kv0 + key) * HD + d0);
            #pragma unroll
            for (int i = 0; i < 16; ++i)
                sVt[(d0 + i) * 32 + key] = bbits(vv[i]);
        }

        // ---- S = Q . K^T  (16x32 = two 16x16 tiles, K=64 over 2 chunks) ----
        v8f s0 = (v8f){0.f,0.f,0.f,0.f,0.f,0.f,0.f,0.f};
        v8f s1 = s0;
        s0 = __builtin_amdgcn_wmma_f32_16x16x32_bf16(
            false, aq[0], false, kb00, (short)0, s0, false, false);
        s1 = __builtin_amdgcn_wmma_f32_16x16x32_bf16(
            false, aq[0], false, kb01, (short)0, s1, false, false);
        s0 = __builtin_amdgcn_wmma_f32_16x16x32_bf16(
            false, aq[1], false, kb10, (short)0, s0, false, false);
        s1 = __builtin_amdgcn_wmma_f32_16x16x32_bf16(
            false, aq[1], false, kb11, (short)0, s1, false, false);

        __syncthreads();   // sVt staged (placed after S WMMAs to overlap)

        // ---- online softmax: row max via DPP16 butterflies (VALU only) ----
        #pragma unroll
        for (int r = 0; r < 8; ++r) {
            float c0 = s0[r], c1 = s1[r];
            float cm = rowmax16(fmaxf(c0, c1));
            const float nm = fmaxf(mrow[r], cm);
            sc8[r] = __expf(mrow[r] - nm);
            mrow[r] = nm;
            const float p0 = __expf(c0 - nm);
            const float p1 = __expf(c1 - nm);
            #pragma unroll
            for (int ns = 0; ns < 4; ++ns) oacc[ns][r] *= sc8[r];
            // P (C-layout) -> LDS row-major [m][key]
            const int prow = (lane & 16) ? r + 8 : r;
            sP[wv][prow * 32 + (lane & 15)]      = f2bu(p0);
            sP[wv][prow * 32 + 16 + (lane & 15)] = f2bu(p1);
        }
        __syncthreads();   // P writes ordered before fragment reads

        // ---- P as A fragment (16 x 32 keys) ----
        const int pm   = lane & 15;
        const int koff = (lane & 16) ? 16 : 0;
        v16bf plo = *reinterpret_cast<const v16bf*>(&sP[wv][pm * 32 + koff]);
        v16bf pa = plo;

        // ---- row sums of P via WMMA against ones (broadcast in C-layout) ----
        v8f lsum = (v8f){0.f,0.f,0.f,0.f,0.f,0.f,0.f,0.f};
        lsum = __builtin_amdgcn_wmma_f32_16x16x32_bf16(
            false, pa, false, ones, (short)0, lsum, false, false);
        #pragma unroll
        for (int r = 0; r < 8; ++r)
            lrow[r] = lrow[r] * sc8[r] + lsum[r];

        // ---- O += P . V : 4 d-subtiles, B frags contiguous in sVt ----
        const int k0 = (lane & 16) ? 16 : 0;
        #pragma unroll
        for (int ns = 0; ns < 4; ++ns) {
            const int n = ns * 16 + (lane & 15);
            v16bf vb = *reinterpret_cast<const v16bf*>(&sVt[n * 32 + k0]);
            oacc[ns] = __builtin_amdgcn_wmma_f32_16x16x32_bf16(
                false, pa, false, vb, (short)0, oacc[ns], false, false);
        }
    }

    // ---- normalize and store merged-head bf16 [b][t][h*64+d] ----
    #pragma unroll
    for (int ns = 0; ns < 4; ++ns) {
        const int d = ns * 16 + (lane & 15);
        #pragma unroll
        for (int r = 0; r < 8; ++r) {
            const int t   = qt * 64 + wv * 16 + ((lane & 16) ? r + 8 : r);
            const float val = oacc[ns][r] / lrow[r];
            o[((size_t)(b * SEQ + t)) * ND + h * HD + d] = f2b(val);
        }
    }
}

// ---------------------------------------------------------------------------
extern "C" void kernel_launch(void* const* d_in, const int* in_sizes, int n_in,
                              void* d_out, int out_size, void* d_ws, size_t ws_size,
                              hipStream_t stream) {
    const float* x  = (const float*)d_in[0];
    const float* Wq = (const float*)d_in[1];
    const float* bq = (const float*)d_in[2];
    const float* Wk = (const float*)d_in[3];
    const float* bk = (const float*)d_in[4];
    const float* Wv = (const float*)d_in[5];
    const float* bv = (const float*)d_in[6];
    const float* Wp = (const float*)d_in[7];
    const float* bp = (const float*)d_in[8];
    float* out = (float*)d_out;

    const size_t QKV_BYTES = (size_t)MROWS * ND * sizeof(__bf16);  // 16 MB
    char* ws   = (char*)d_ws;
    __bf16* qbuf = (__bf16*)(ws);
    __bf16* kbuf = (__bf16*)(ws + QKV_BYTES);
    __bf16* vbuf = (__bf16*)(ws + 2 * QKV_BYTES);
    __bf16* obuf = (__bf16*)(ws + 3 * QKV_BYTES);

    dim3 gblk(128);
    dim3 ggrid(ND / 64, MROWS / 64);   // (16, 128)

    // Q/K/V projections -> bf16, head-sliced [h][b][t][64]
    gemm_wmma<true,  true ><<<ggrid, gblk, 0, stream>>>(x, Wq, bq, qbuf);
    gemm_wmma<true,  true ><<<ggrid, gblk, 0, stream>>>(x, Wk, bk, kbuf);
    gemm_wmma<true,  true ><<<ggrid, gblk, 0, stream>>>(x, Wv, bv, vbuf);

    // attention -> bf16 merged heads [b][t][1024]
    dim3 agrid(SEQ / 64, BATCH, NH);   // (32, 4, 16)
    attn_wmma<<<agrid, dim3(128), 0, stream>>>(qbuf, kbuf, vbuf, obuf);

    // output projection -> fp32 d_out [8192][1024]
    gemm_wmma<false, false><<<ggrid, gblk, 0, stream>>>(obuf, Wp, bp, out);
}
